// QuantizedExpert_59657095741907
// MI455X (gfx1250) — matrix operand order repaired and checked by reading, not compiled, not run on hardware
//
#include <hip/hip_runtime.h>

// ---------------------------------------------------------------------------
// Quantized FFN: out = gelu(X @ dequant4(W1).T) @ dequant4(W2).T
// bf16 WMMA GEMMs, double-buffered LDS: A tiles staged by the Tensor Data
// Mover (async DMA) + int4 weights dequantized in-flight, overlapped with the
// WMMA compute on the other buffer. One barrier per K-step.
// ---------------------------------------------------------------------------

typedef __attribute__((ext_vector_type(16))) __bf16 v16bf;
typedef __attribute__((ext_vector_type(8)))  __bf16 v8bf;
typedef __attribute__((ext_vector_type(8)))  float  v8f;
typedef __attribute__((ext_vector_type(4)))  unsigned int v4u;
typedef __attribute__((ext_vector_type(8)))  int    v8i;
typedef __attribute__((ext_vector_type(4)))  int    v4i;

#define D_MODEL  4096
#define HIDDEN   16384
#define N_TOKENS 4096

#define BM 128            // block tile rows (tokens)
#define BN 128            // block tile cols (output features)
#define BK 32             // K per WMMA (bf16)
#define LDSS 40           // BK + 8 bf16 pad -> 80B row stride (16B aligned)

#if __has_builtin(__builtin_amdgcn_tensor_load_to_lds)
#define USE_TDM 1
#if __has_include(<hip/amd_detail/amd_gfx1250_TDM.h>)
#define TDM_ARITY6 1
#endif
#endif

__device__ __forceinline__ float gelu_f(float x) {
    const float c0 = 0.7978845608028654f;   // sqrt(2/pi)
    const float c1 = 0.044715f;
    float x3 = x * x * x;
    return 0.5f * x * (1.0f + tanhf(c0 * (x + c1 * x3)));
}

__device__ __forceinline__ v16bf concat8(v8bf a, v8bf b) {
    return __builtin_shufflevector(a, b, 0, 1, 2, 3, 4, 5, 6, 7,
                                         8, 9, 10, 11, 12, 13, 14, 15);
}

#ifdef USE_TDM
// One TDM 2D tile load: rows x cols bf16 tile at gtile (row stride in elems)
// into LDS at byte offset lds_off; pads each 64B row by 16B -> 80B (LDSS).
__device__ __forceinline__ void tdm_load_tile_bf16(const __bf16* gtile,
                                                   unsigned lds_off,
                                                   int rows, int cols,
                                                   int row_stride_elems) {
    unsigned long long ga = (unsigned long long)(const void*)gtile;
    v4u g0;
    g0[0] = 1u;                                   // count=1 (valid user D#)
    g0[1] = lds_off;                              // lds_addr (bytes)
    g0[2] = (unsigned)(ga & 0xffffffffu);         // global_addr[31:0]
    g0[3] = (unsigned)((ga >> 32) & 0x01ffffffu)  // global_addr[56:32]
            | (2u << 30);                         // type=2 ("image")
    unsigned td0 = (unsigned)row_stride_elems;    // tensor_dim0 (elements)
    unsigned td1 = 0x40000000u;                   // tensor_dim1: huge (no OOB)
    unsigned long long st0 = (unsigned long long)row_stride_elems;
    v8i g1;
    g1[0] = (int)((1u << 16)      // data_size = 2 bytes
                | (1u << 20)      // pad_enable
                | (3u << 22)      // pad_interval: 16 DWORDs (one 64B row)
                | (3u << 25));    // pad_amount: 4 DWORDs (16B) -> 80B stride
    g1[1] = (int)((td0 & 0xffffu) << 16);                       // td0[15:0]
    g1[2] = (int)((td0 >> 16) | ((td1 & 0xffffu) << 16));       // td0 hi|td1 lo
    g1[3] = (int)((td1 >> 16) | ((unsigned)cols << 16));        // tile_dim0
    g1[4] = (int)((unsigned)rows & 0xffffu);                    // tile_dim1
    g1[5] = (int)(st0 & 0xffffffffu);                           // dim0 stride
    g1[6] = (int)((st0 >> 32) & 0xffffu);
    g1[7] = 0;
    v4i z4 = {0, 0, 0, 0};
#ifdef TDM_ARITY6
    v8i z8 = {0, 0, 0, 0, 0, 0, 0, 0};
    __builtin_amdgcn_tensor_load_to_lds(g0, g1, z4, z4, z8, 0);
#else
    __builtin_amdgcn_tensor_load_to_lds(g0, g1, z4, z4, 0);
#endif
}
#endif

__device__ __forceinline__ void wait_tensor0() {
#if __has_builtin(__builtin_amdgcn_s_wait_tensorcnt)
    __builtin_amdgcn_s_wait_tensorcnt(0);
#elif defined(USE_TDM)
    asm volatile("s_wait_tensorcnt 0x0" ::: "memory");
#endif
}

// ---------------------------------------------------------------------------
// X f32 -> bf16 pre-conversion (makes GEMM1's A tile a pure byte move)
// ---------------------------------------------------------------------------
__global__ __launch_bounds__(256)
void x_to_bf16(const float* __restrict__ in, __bf16* __restrict__ out) {
    size_t i = ((size_t)blockIdx.x * 256 + threadIdx.x) * 8;
    float4 a = *(const float4*)(in + i);
    float4 b = *(const float4*)(in + i + 4);
    v8bf o;
    o[0] = (__bf16)a.x; o[1] = (__bf16)a.y; o[2] = (__bf16)a.z; o[3] = (__bf16)a.w;
    o[4] = (__bf16)b.x; o[5] = (__bf16)b.y; o[6] = (__bf16)b.z; o[7] = (__bf16)b.w;
    *(v8bf*)(out + i) = o;
}

// ---------------------------------------------------------------------------
// Tiled GEMM: C[M,N] = scale[n] * (A[M,K]bf16 . dequant4(W[N,K/2]).T) (+gelu)
// 256 threads = 8 wave32 (2x4); wave tile 64x32 -> 8 f32 accumulators.
// ---------------------------------------------------------------------------
template <bool FUSE_GELU>
__global__ __launch_bounds__(256)
void ffn_int4_wmma(const __bf16* __restrict__ A,
                   const int*    __restrict__ Wpk,     // one byte per int32
                   const float*  __restrict__ scales,  // [N]
                   void*         __restrict__ Cptr,    // bf16 (gelu) or f32
                   int M, int N, int K)
{
    __shared__ __bf16 Asm[2][BM][LDSS];   // double-buffered activation tile
    __shared__ __bf16 Bsm[2][BN][LDSS];   // double-buffered dequant W tile

    const int tid   = threadIdx.x;
    const int lane  = tid & 31;
    const int wave  = tid >> 5;        // 0..7
    const int waveM = wave & 1;
    const int waveN = wave >> 1;

    const int n0 = blockIdx.x * BN;
    const int m0 = blockIdx.y * BM;

    const int col16 = lane & 15;
    const int half  = lane >> 4;

    const int brow = tid >> 1;         // 0..127
    const int bj0  = (tid & 1) * 8;    // packed-int start within the 16/row
#ifndef USE_TDM
    const int arow = tid >> 1;
    const int acol = (tid & 1) * 16;
#endif

    const int KP = K >> 1;
    const int NK = K / BK;

#ifdef USE_TDM
    // LDS byte address = low 32 bits of the flat pointer
    const unsigned asm_lds0 = (unsigned)(unsigned long long)(void*)&Asm[0][0][0];
    const unsigned asm_ldsz = (unsigned)(BM * LDSS * sizeof(__bf16));
#endif

    v8f acc[4][2] = {};

    // ---- tile stager: A via TDM (wave 0), B via dequant into LDS ----------
    auto stage = [&](int it, int buf) {
        const int k0 = it * BK;
#ifdef USE_TDM
        if (wave == 0) {
            tdm_load_tile_bf16(A + (size_t)m0 * K + k0,
                               asm_lds0 + (unsigned)buf * asm_ldsz, BM, BK, K);
        }
#else
        {
            const __bf16* g = A + (size_t)(m0 + arow) * K + k0 + acol;
            v8bf a0 = *(const v8bf*)g;
            v8bf a1 = *(const v8bf*)(g + 8);
            *(v8bf*)&Asm[buf][arow][acol]     = a0;
            *(v8bf*)&Asm[buf][arow][acol + 8] = a1;
        }
#endif
        const int* gp = Wpk + (size_t)(n0 + brow) * KP + (k0 >> 1) + bj0;
        int4 p0 = *(const int4*)gp;
        int4 p1 = *(const int4*)(gp + 4);
        __builtin_prefetch(gp + (BK >> 1), 0, 0);      // one staged tile ahead
        int v[8] = { p0.x, p0.y, p0.z, p0.w, p1.x, p1.y, p1.z, p1.w };
#pragma unroll
        for (int j = 0; j < 8; ++j) {
            int b = v[j];
            // high nibble = even column, low nibble = odd column
            Bsm[buf][brow][2 * (bj0 + j) + 0] = (__bf16)((b >> 4) - 8);
            Bsm[buf][brow][2 * (bj0 + j) + 1] = (__bf16)((b & 15) - 8);
        }
    };

    stage(0, 0);                                       // prologue

    for (int it = 0; it < NK; ++it) {
        const int cur = it & 1;

        wait_tensor0();            // TDM tile(it) landed (issuing wave)
        __syncthreads();           // tile(it) visible; buffer cur^1 is free

        if (it + 1 < NK) stage(it + 1, cur ^ 1);       // overlap with compute

        // ------------- fragments (ISA bf16 A/B VGPR layouts) ---------------
        v16bf afr[4];
#pragma unroll
        for (int mi = 0; mi < 4; ++mi) {
            int m = waveM * 64 + mi * 16 + col16;
            afr[mi] = concat8(*(const v8bf*)&Asm[cur][m][half * 8],
                              *(const v8bf*)&Asm[cur][m][16 + half * 8]);
        }
        v16bf bfr[2];
#pragma unroll
        for (int ni = 0; ni < 2; ++ni) {
            int n = waveN * 32 + ni * 16 + col16;
            bfr[ni] = concat8(*(const v8bf*)&Bsm[cur][n][half * 8],
                              *(const v8bf*)&Bsm[cur][n][16 + half * 8]);
        }

#pragma unroll
        for (int mi = 0; mi < 4; ++mi)
#pragma unroll
            for (int ni = 0; ni < 2; ++ni)
                acc[mi][ni] = __builtin_amdgcn_wmma_f32_16x16x32_bf16(
                    false, afr[mi], false, bfr[ni],
                    (short)0, acc[mi][ni], false, false);
    }

    // ------- epilogue: per-column scale (+gelu); C/D: vgpr i -> M = i + 8*half
#pragma unroll
    for (int ni = 0; ni < 2; ++ni) {
        int ng = n0 + waveN * 32 + ni * 16 + col16;
        float s = scales[ng];
#pragma unroll
        for (int mi = 0; mi < 4; ++mi) {
#pragma unroll
            for (int i = 0; i < 8; ++i) {
                int mg = m0 + waveM * 64 + mi * 16 + half * 8 + i;
                float val = acc[mi][ni][i] * s;
                if (FUSE_GELU) {
                    ((__bf16*)Cptr)[(size_t)mg * N + ng] = (__bf16)gelu_f(val);
                } else {
                    ((float*)Cptr)[(size_t)mg * N + ng] = val;
                }
            }
        }
    }
}

extern "C" void kernel_launch(void* const* d_in, const int* in_sizes, int n_in,
                              void* d_out, int out_size, void* d_ws, size_t ws_size,
                              hipStream_t stream) {
    (void)in_sizes; (void)n_in; (void)out_size; (void)ws_size;

    const float* x   = (const float*)d_in[0];  // [N_TOKENS, D_MODEL] f32
    const int*   w1p = (const int*)d_in[1];    // [HIDDEN, D_MODEL/2]
    const float* w1s = (const float*)d_in[2];  // [HIDDEN]
    const int*   w2p = (const int*)d_in[3];    // [D_MODEL, HIDDEN/2]
    const float* w2s = (const float*)d_in[4];  // [D_MODEL]
    float*       out = (float*)d_out;          // [N_TOKENS, D_MODEL] f32

    // workspace: h bf16 [N_TOKENS, HIDDEN] then Xbf16 [N_TOKENS, D_MODEL]
    __bf16* h  = (__bf16*)d_ws;
    __bf16* xb = (__bf16*)((char*)d_ws + (size_t)N_TOKENS * HIDDEN * 2);

    dim3 blk(256);

    // 0) X -> bf16 (16.7M elems, 8/thread)
    x_to_bf16<<<dim3((N_TOKENS * (size_t)D_MODEL) / (256 * 8)), blk, 0, stream>>>(x, xb);

    // 1) h = gelu(scale1 * (Xbf16 . W1int4^T))
    ffn_int4_wmma<true><<<dim3(HIDDEN / BN, N_TOKENS / BM), blk, 0, stream>>>(
        xb, w1p, w1s, h, N_TOKENS, HIDDEN, D_MODEL);

    // 2) out = scale2 * (h . W2int4^T)
    ffn_int4_wmma<false><<<dim3(D_MODEL / BN, N_TOKENS / BM), blk, 0, stream>>>(
        h, w2p, w2s, out, N_TOKENS, D_MODEL, HIDDEN);
}
